// ODEEulerSolver_37460704755843
// MI455X (gfx1250) — matrix operand order repaired and checked by reading, not compiled, not run on hardware
//
#include <hip/hip_runtime.h>

// ---------------------------------------------------------------------------
// ODE Euler scan: z_{t+1} = z_t + W2^T tanh(W1^T z_t + b1) + b2, with per-batch
// teacher forcing. B=64 C=256 T=64 V=25 H=512.
// 1600 independent (b,v) columns -> 100 workgroups x 16 columns, each WG runs
// the full 63-step scan with v_wmma_f32_16x16x32_bf16 (bf16 in, f32 acc).
// Weight A-tiles are loop-invariant and end up register-resident (compiler
// LICM across the t-loop, verified in disasm via s_set_vgpr_msb high-VGPR use).
// ---------------------------------------------------------------------------

typedef __attribute__((ext_vector_type(16))) __bf16 v16bf;
typedef __attribute__((ext_vector_type(8)))  float  v8f;

static constexpr int Bn = 64;
static constexpr int Cc = 256;
static constexpr int Tt = 64;
static constexpr int Vv = 25;
static constexpr int Hh = 512;

// LDS row strides (padded so lane addresses land 4 banks apart)
static constexpr int ZS_STRIDE = 260;  // f32
static constexpr int ZB_STRIDE = 264;  // bf16 (even -> dword-pair loads aligned)
static constexpr int US_STRIDE = 520;  // bf16

__device__ __forceinline__ ushort f2bf(float f) {
    // round-to-nearest-even f32 -> bf16
    uint u = __float_as_uint(f);
    u += 0x7fffu + ((u >> 16) & 1u);
    return (ushort)(u >> 16);
}

// Branch-free tanh: prefer hardware v_tanh_f32 (gfx1250 TRANS op); otherwise
// tanh(x) = copysign((1-e)/(1+e), x), e = 2^(-2*log2(e)*|x|)  (v_exp + v_rcp).
__device__ __forceinline__ float tanh_fast(float x) {
#if __has_builtin(__builtin_amdgcn_tanhf)
    return __builtin_amdgcn_tanhf(x);
#else
    float ax = __builtin_fabsf(x);
    float e  = __builtin_amdgcn_exp2f(ax * -2.885390081777927f); // e^(-2ax)
    float r  = (1.0f - e) * __builtin_amdgcn_rcpf(1.0f + e);
    return __builtin_copysignf(r, x);
#endif
}

// A-operand: pre-packed bf16 tiles, lane-major: ((tile*32 + lane)*16 + e)
__device__ __forceinline__ v16bf load_a(const ushort* __restrict__ Wp, int tile, int lane) {
    union { uint4 q[2]; v16bf v; } x;
    const uint4* p = (const uint4*)(Wp + (size_t)(tile * 32 + lane) * 16);
    x.q[0] = p[0];
    x.q[1] = p[1];
    return x.v;
}

// B-operand (32x16 bf16) gathered from LDS stored as [n][k] (k contiguous).
// Order used here: lane L: n = L%16, kh = L/16; VGPR g holds k = kh*16 + 2g, 2g+1.
// 8 contiguous dwords -> compiler merges into two ds_load_b128.
__device__ __forceinline__ v16bf load_b(const ushort* sh, int stride, int kbase,
                                        int n, int kh) {
    union { uint u[8]; v16bf v; } x;
    const ushort* base = sh + n * stride + kbase + kh * 16;
#pragma unroll
    for (int g = 0; g < 8; ++g)
        x.u[g] = *(const uint*)(base + g * 2);
    return x.v;
}

// ---------------------------------------------------------------------------
// One-shot weight repack: W1^T (HxC) and W2^T (CxH) into bf16 WMMA A-operand
// order. 16-bit A 16x32 order used here: lane m = L%16, half = L/16,
// element e -> k = half*16 + e (contiguous, matches load_b).
// ---------------------------------------------------------------------------
__global__ void ode_pack_weights(const float* __restrict__ W1,
                                 const float* __restrict__ W2,
                                 ushort* __restrict__ W1p,
                                 ushort* __restrict__ W2p) {
    int tid   = blockIdx.x * blockDim.x + threadIdx.x;   // 262144 total
    int which = tid >> 17;          // 131072 elements each
    int i     = tid & 0x1FFFF;
    int e     = i & 15;
    int lane  = (i >> 4) & 31;
    int tile  = i >> 9;
    int m = lane & 15, half = lane >> 4;
    int g = e >> 1,  pos  = e & 1;
    int k = half * 16 + g * 2 + pos;
    if (which == 0) {
        // GEMM1 A = W1^T : M = H (32 tiles), K = C (8 chunks); tile = mt*8 + kt
        int kt = tile & 7, mt = tile >> 3;
        int hm = mt * 16 + m;
        int kc = kt * 32 + k;
        W1p[(size_t)(tile * 32 + lane) * 16 + e] = f2bf(W1[kc * Hh + hm]);
    } else {
        // GEMM2 A = W2^T : M = C (16 tiles), K = H (16 chunks); tile = mt*16 + kt
        int kt = tile & 15, mt = tile >> 4;
        int cm = mt * 16 + m;
        int kh = kt * 32 + k;
        W2p[(size_t)(tile * 32 + lane) * 16 + e] = f2bf(W2[kh * Cc + cm]);
    }
}

// ---------------------------------------------------------------------------
// Main persistent-scan kernel: 100 blocks x 256 threads (8 wave32).
// ---------------------------------------------------------------------------
__global__ __launch_bounds__(256, 1)
void ode_scan(const float* __restrict__ h,
              const float* __restrict__ b1,
              const float* __restrict__ b2,
              const float* __restrict__ tf_rand,
              const ushort* __restrict__ W1p,
              const ushort* __restrict__ W2p,
              float* __restrict__ out) {
    __shared__ float  Zs[16 * ZS_STRIDE];   // f32 state z_t     [n][c]
    __shared__ ushort Zb[16 * ZB_STRIDE];   // bf16 of z_t       [n][c]
    __shared__ ushort Us[16 * US_STRIDE];   // bf16 hidden u     [n][h]
    __shared__ float  b1s[Hh];
    __shared__ float  b2s[Cc];
    __shared__ int    bcol[16];
    __shared__ int    vcol[16];

    const int tid  = threadIdx.x;
    const int lane = tid & 31;
    const int wave = tid >> 5;
    const int n    = lane & 15;      // column within tile (C/D matrix N index)
    const int kh   = lane >> 4;      // lane half
    const int g0   = blockIdx.x * 16;

    if (tid < 16) {
        int col = g0 + tid;          // flattened (b, v)
        bcol[tid] = col / Vv;
        vcol[tid] = col % Vv;
    }
    for (int i = tid; i < Hh; i += 256) b1s[i] = b1[i];
    for (int i = tid; i < Cc; i += 256) b2s[i] = b2[i];
    __syncthreads();

    // ---- init: z0 = h[:, :, 0, :]; also emit output slice 0 ----
    {
        int j = tid & 15, ci = tid >> 4;
        int b = bcol[j], v = vcol[j];
#pragma unroll 4
        for (int cc = 0; cc < Cc; cc += 16) {
            int c = cc + ci;
            float z = h[((b * Cc + c) * Tt + 0) * Vv + v];
            Zs[j * ZS_STRIDE + c] = z;
            out[((b * Cc + c) * Tt + 0) * Vv + v] = z;
        }
    }
    __syncthreads();

    for (int t = 0; t < Tt - 1; ++t) {
        // ---- teacher-forcing select + bf16 convert ----
        {
            int j = tid & 15, ci = tid >> 4;
            int b = bcol[j], v = vcol[j];
            // t==0: h slice 0 == z0, select is identity either way
            bool take_h = (t > 0) && !(tf_rand[t * Bn + b] < 0.5f);
#pragma unroll 4
            for (int cc = 0; cc < Cc; cc += 16) {
                int c = cc + ci;
                float z;
                if (take_h) {
                    z = h[((b * Cc + c) * Tt + t) * Vv + v];
                    Zs[j * ZS_STRIDE + c] = z;
                } else {
                    z = Zs[j * ZS_STRIDE + c];
                }
                Zb[j * ZB_STRIDE + c] = f2bf(z);
            }
        }
        __syncthreads();

        // ---- GEMM1: U = tanh(W1^T z_t + b1), H=512 -> 32 M-tiles / 8 waves ----
        for (int mi = 0; mi < 4; ++mi) {
            int mt = wave * 4 + mi;
            v8f acc = {};
#pragma unroll
            for (int kt = 0; kt < 8; ++kt) {
                v16bf a = load_a(W1p, mt * 8 + kt, lane);
                v16bf bm = load_b(Zb, ZB_STRIDE, kt * 32, n, kh);
                acc = __builtin_amdgcn_wmma_f32_16x16x32_bf16(
                          false, a, false, bm, (short)0, acc, false, false);
            }
            // D layout: lane n = L%16; element r -> m = r + 8*(L/16)
            int mbase = mt * 16 + kh * 8;
#pragma unroll
            for (int r = 0; r < 8; r += 2) {
                float x0 = tanh_fast(acc[r]     + b1s[mbase + r]);
                float x1 = tanh_fast(acc[r + 1] + b1s[mbase + r + 1]);
                uint pk = (uint)f2bf(x0) | ((uint)f2bf(x1) << 16);
                *(uint*)&Us[n * US_STRIDE + mbase + r] = pk;
            }
        }
        __syncthreads();

        // ---- GEMM2 + Euler update: z_{t+1} = z_t + W2^T U + b2 ----
        for (int mi = 0; mi < 2; ++mi) {
            int mt = wave * 2 + mi;
            v8f acc = {};
#pragma unroll
            for (int kt = 0; kt < 16; ++kt) {
                v16bf a = load_a(W2p, mt * 16 + kt, lane);
                v16bf bm = load_b(Us, US_STRIDE, kt * 32, n, kh);
                acc = __builtin_amdgcn_wmma_f32_16x16x32_bf16(
                          false, a, false, bm, (short)0, acc, false, false);
            }
            int cbase = mt * 16 + kh * 8;
            int b = bcol[n], v = vcol[n];
#pragma unroll
            for (int r = 0; r < 8; ++r) {
                int c = cbase + r;
                float z = Zs[n * ZS_STRIDE + c] + acc[r] + b2s[c];
                Zs[n * ZS_STRIDE + c] = z;
                out[((b * Cc + c) * Tt + (t + 1)) * Vv + v] = z;
            }
        }
        __syncthreads();
    }
}

// ---------------------------------------------------------------------------
extern "C" void kernel_launch(void* const* d_in, const int* in_sizes, int n_in,
                              void* d_out, int out_size, void* d_ws, size_t ws_size,
                              hipStream_t stream) {
    const float* h  = (const float*)d_in[0];
    const float* W1 = (const float*)d_in[1];
    const float* b1 = (const float*)d_in[2];
    const float* W2 = (const float*)d_in[3];
    const float* b2 = (const float*)d_in[4];
    const float* tf = (const float*)d_in[5];
    // d_in[6] = t_obs (unused by the reference computation)
    float* out = (float*)d_out;

    // workspace: packed bf16 weights (256 KB each)
    ushort* W1p = (ushort*)d_ws;
    ushort* W2p = W1p + (size_t)Hh * Cc;

    ode_pack_weights<<<(2 * Hh * Cc) / 256, 256, 0, stream>>>(W1, W2, W1p, W2p);
    ode_scan<<<(Bn * Vv) / 16, 256, 0, stream>>>(h, b1, b2, tf, W1p, W2p, out);
}
